// CausalSelfAttention_71150428225628
// MI455X (gfx1250) — matrix operand order repaired and checked by reading, not compiled, not run on hardware
//
#include <hip/hip_runtime.h>

typedef __bf16 bf16;
typedef __attribute__((ext_vector_type(16))) __bf16 v16bf;
typedef __attribute__((ext_vector_type(8)))  __bf16 v8bf;
typedef __attribute__((ext_vector_type(8)))  float   v8f;
typedef __attribute__((ext_vector_type(4)))  unsigned int v4u;
typedef __attribute__((ext_vector_type(8)))  int     v8i;
typedef __attribute__((ext_vector_type(4)))  int     v4i;

#define NHEAD 16
#define NKVH  4
#define SEQ   2048
#define DM    2048
#define HDIM  128
#define BATCH 2

// ---------------------------------------------------------------------------
// Tensor Data Mover helpers (CDNA5 TDM, ISA cdna5 ch.7/8 of 08_async_tensor)
// D# group0: [1:0]=count, [63:32]=lds_addr, [120:64]=global_addr, [127:126]=type(2)
// D# group1: [17:16]=data_size, [79:48]=tensor_dim0, [111:80]=tensor_dim1,
//            [127:112]=tile_dim0, [143:128]=tile_dim1, [207:160]=dim0_stride
// ---------------------------------------------------------------------------
__device__ __forceinline__ unsigned int lds_off(const void* p) {
  // low 32 bits of a flat LDS-aperture address are the LDS byte address
  return (unsigned int)(unsigned long long)(uintptr_t)p;
}

__device__ __forceinline__ void tdm_issue(v4u g0, v8i g1) {
  v4i z4 = (v4i){0, 0, 0, 0};
#if defined(__clang_major__) && __clang_major__ >= 23
  v8i z8 = (v8i){0, 0, 0, 0, 0, 0, 0, 0};
  __builtin_amdgcn_tensor_load_to_lds(g0, g1, z4, z4, z8, 0);
#else
  __builtin_amdgcn_tensor_load_to_lds(g0, g1, z4, z4, 0);
#endif
}

// 2D bf16 tile load: tile_d0 elems along contiguous dim, tile_d1 rows,
// row stride (elements) = stride_e. Dest LDS: packed row-major tile.
__device__ __forceinline__ void tdm_load_2d(const void* gsrc, unsigned int lds_byte,
                                            unsigned int tile_d0, unsigned int tile_d1,
                                            unsigned long long stride_e) {
  unsigned long long ga = (unsigned long long)(uintptr_t)gsrc;
  v4u g0;
  g0.x = 1u;                                            // count=1, user mode
  g0.y = lds_byte;                                      // lds_addr
  g0.z = (unsigned int)(ga & 0xffffffffu);              // global_addr[31:0]
  g0.w = (unsigned int)((ga >> 32) & 0x01ffffffu) | (2u << 30);  // addr[56:32] | type=2
  v8i g1;
  g1[0] = (int)(1u << 16);                              // data_size=1 -> 2 bytes
  g1[1] = (int)((tile_d0 & 0xffffu) << 16);             // tensor_dim0[15:0] (== tile_d0)
  g1[2] = (int)((tile_d0 >> 16) | ((tile_d1 & 0xffffu) << 16));  // td0 hi | tensor_dim1 lo
  g1[3] = (int)((tile_d1 >> 16) | (tile_d0 << 16));     // tensor_dim1 hi | tile_dim0
  g1[4] = (int)(tile_d1 & 0xffffu);                     // tile_dim1 (tile_dim2 = 0)
  g1[5] = (int)(stride_e & 0xffffffffu);                // tensor_dim0_stride lo
  g1[6] = (int)((stride_e >> 32) & 0xffffu);            // tensor_dim0_stride hi
  g1[7] = 0;
  tdm_issue(g0, g1);
}

// ---------------------------------------------------------------------------
// WMMA fragment loaders (CDNA5 wave32 layouts, 16x16x32 bf16)
// ---------------------------------------------------------------------------
__device__ __forceinline__ v16bf frag_a(const bf16* base, int lda, int lane, int koff) {
  int row  = lane & 15;
  int half = lane >> 4;
  const bf16* p = base + row * lda + koff;
  v8bf lo = *(const v8bf*)(p + half * 16);
  v8bf hi = *(const v8bf*)(p + half * 16 + 8);
  return __builtin_shufflevector(lo, hi, 0,1,2,3,4,5,6,7,8,9,10,11,12,13,14,15);
}

__device__ __forceinline__ v16bf frag_b(const bf16* base, int ldk, int lane, int nbase, int koff) {
  int n    = lane & 15;
  int half = lane >> 4;
  return *(const v16bf*)(base + (nbase + n) * ldk + koff + half * 16);
}

// ---------------------------------------------------------------------------
// Kernel 1: W_eff[n][k] = dequant_i8(W[n][k]) + (A B)^T   (bf16, [N][K])
// ---------------------------------------------------------------------------
__global__ __launch_bounds__(256)
void prep_weight(const float* __restrict__ W, const float* __restrict__ Alo,
                 const float* __restrict__ Blo, bf16* __restrict__ Out,
                 int nrows, int K, int row_off) {
  int n = blockIdx.x;
  int tid = threadIdx.x;
  __shared__ float red[8];
  __shared__ float bcol[16];

  float mx = 0.f;
  for (int k = tid; k < K; k += 256) mx = fmaxf(mx, fabsf(W[(size_t)n * K + k]));
#pragma unroll
  for (int m = 1; m < 32; m <<= 1) mx = fmaxf(mx, __shfl_xor(mx, m, 32));
  if ((tid & 31) == 0) red[tid >> 5] = mx;
  if (tid < 16) bcol[tid] = Blo[(size_t)tid * nrows + n];
  __syncthreads();
  float gmax = 0.f;
#pragma unroll
  for (int i = 0; i < 8; ++i) gmax = fmaxf(gmax, red[i]);
  float scale = fmaxf(gmax * (1.f / 127.f), 1.f / 127.f);

  for (int k = tid; k < K; k += 256) {
    float lora = 0.f;
    const float* ar = Alo + (size_t)k * 16;
#pragma unroll
    for (int r = 0; r < 16; ++r) lora += ar[r] * bcol[r];
    float w  = W[(size_t)n * K + k];
    float wd = fminf(fmaxf(rintf(w / scale), -127.f), 127.f) * scale;
    Out[(size_t)(row_off + n) * K + k] = (bf16)(wd + lora);
  }
}

// ---------------------------------------------------------------------------
// Kernel 2/5: tiled bf16 WMMA GEMM.  C[M x N] = A[M x K] * Bw[N x K]^T
// 256 threads (8 waves), 128x128 tile, BK=32. B (and bf16 A) tiles staged by
// TDM (tensor_load_to_lds) with double buffering; f32 A converted manually.
// ---------------------------------------------------------------------------
template <bool A_IS_BF16>
__global__ __launch_bounds__(256)
void gemm_wmma(const void* __restrict__ Aptr, const bf16* __restrict__ Bw,
               float* __restrict__ C, int K, int ldc) {
  __shared__ __align__(32) bf16 As[2][128 * 32];
  __shared__ __align__(32) bf16 Bs[2][128 * 32];

  int tid  = threadIdx.x;
  int lane = tid & 31;
  int wave = tid >> 5;
  int bm = blockIdx.y * 128;
  int bn = blockIdx.x * 128;
  int wm = (wave & 3) * 32;
  int wn = (wave >> 2) * 64;

  const bf16* Ab = (const bf16*)Aptr;
  const float* Af = (const float*)Aptr;

  v8f acc[2][4];
#pragma unroll
  for (int i = 0; i < 2; ++i)
#pragma unroll
    for (int j = 0; j < 4; ++j) acc[i][j] = (v8f){};

  int lrow = tid >> 1;
  int lcs  = (tid & 1) * 16;

  // prologue: wave0 launches TDM for the first tiles into buffer 0
  if (wave == 0) {
    tdm_load_2d(Bw + (size_t)bn * K, lds_off(&Bs[0][0]), 32, 128, (unsigned long long)K);
    if (A_IS_BF16)
      tdm_load_2d(Ab + (size_t)bm * K, lds_off(&As[0][0]), 32, 128, (unsigned long long)K);
  }

  int buf = 0;
  for (int k0 = 0; k0 < K; k0 += 32) {
    float4 fa[4];
    if (!A_IS_BF16) {
      const float* p = Af + (size_t)(bm + lrow) * K + k0 + lcs;
#pragma unroll
      for (int i = 0; i < 4; ++i) fa[i] = *(const float4*)(p + i * 4);
    }
    __syncthreads();  // previous compute finished: safe to overwrite buffers
    if (!A_IS_BF16) {
#pragma unroll
      for (int i = 0; i < 4; ++i) {
        As[0][lrow * 32 + lcs + i * 4 + 0] = (bf16)fa[i].x;
        As[0][lrow * 32 + lcs + i * 4 + 1] = (bf16)fa[i].y;
        As[0][lrow * 32 + lcs + i * 4 + 2] = (bf16)fa[i].z;
        As[0][lrow * 32 + lcs + i * 4 + 3] = (bf16)fa[i].w;
      }
    }
    if (wave == 0) {
      if (k0 + 32 < K) {  // issue next tiles into the other buffer
        tdm_load_2d(Bw + (size_t)bn * K + k0 + 32, lds_off(&Bs[buf ^ 1][0]),
                    32, 128, (unsigned long long)K);
        if (A_IS_BF16)
          tdm_load_2d(Ab + (size_t)bm * K + k0 + 32, lds_off(&As[buf ^ 1][0]),
                      32, 128, (unsigned long long)K);
        __builtin_amdgcn_s_wait_tensorcnt(A_IS_BF16 ? 2 : 1);  // current tiles done
      } else {
        __builtin_amdgcn_s_wait_tensorcnt(0);
      }
    }
    __syncthreads();  // staged data visible to all waves

    const bf16* Ause = A_IS_BF16 ? &As[buf][0] : &As[0][0];
    v16bf af[2], bfr[4];
#pragma unroll
    for (int mt = 0; mt < 2; ++mt) af[mt] = frag_a(Ause + (wm + mt * 16) * 32, 32, lane, 0);
#pragma unroll
    for (int nt = 0; nt < 4; ++nt) bfr[nt] = frag_b(&Bs[buf][0], 32, lane, wn + nt * 16, 0);
#pragma unroll
    for (int mt = 0; mt < 2; ++mt)
#pragma unroll
      for (int nt = 0; nt < 4; ++nt)
        acc[mt][nt] = __builtin_amdgcn_wmma_f32_16x16x32_bf16(
            false, af[mt], false, bfr[nt], (short)0, acc[mt][nt], false, false);
    buf ^= 1;
  }

  int half = lane >> 4;
  int cn   = lane & 15;
#pragma unroll
  for (int mt = 0; mt < 2; ++mt)
#pragma unroll
    for (int nt = 0; nt < 4; ++nt)
#pragma unroll
      for (int r = 0; r < 8; ++r) {
        int row = bm + wm + mt * 16 + r + half * 8;
        int col = bn + wn + nt * 16 + cn;
        C[(size_t)row * ldc + col] = acc[mt][nt][r];
      }
}

// ---------------------------------------------------------------------------
// Kernel 3: per-head RMS-norm + RoPE + gain, f32 QKV -> bf16 head-major
// ---------------------------------------------------------------------------
__global__ __launch_bounds__(128)
void normrope(const float* __restrict__ QKV, const float* __restrict__ qgain,
              bf16* __restrict__ Qh, bf16* __restrict__ Kh, bf16* __restrict__ Vh) {
  const int NQ = BATCH * NHEAD * SEQ;
  const int NK = BATCH * NKVH * SEQ;
  int idx = blockIdx.x;
  int tid = threadIdx.x;

  int kind, b, h, s, coloff;
  if (idx < NQ) {
    kind = 0; b = idx >> 15; h = (idx >> 11) & 15; s = idx & 2047;
    coloff = h * HDIM;
  } else if (idx < NQ + NK) {
    int j = idx - NQ;
    kind = 1; b = j >> 13; h = (j >> 11) & 3; s = j & 2047;
    coloff = DM + h * HDIM;
  } else {
    int j = idx - NQ - NK;
    kind = 2; b = j >> 13; h = (j >> 11) & 3; s = j & 2047;
    coloff = DM + NKVH * HDIM + h * HDIM;
  }

  float x = QKV[(size_t)(b * SEQ + s) * 3072 + coloff + tid];

  if (kind < 2) {
    __shared__ float red[4];
    __shared__ float buf[128];
    float ss = x * x;
#pragma unroll
    for (int m = 1; m < 32; m <<= 1) ss += __shfl_xor(ss, m, 32);
    if ((tid & 31) == 0) red[tid >> 5] = ss;
    __syncthreads();
    float tot = red[0] + red[1] + red[2] + red[3];
    float nrm = x * rsqrtf(tot * (1.f / 128.f) + 1.1920929e-07f);
    buf[tid] = nrm;
    __syncthreads();

    float outv;
    if (tid < 32) {
      float ang = (float)s * __powf(10000.f, -(float)tid * (1.f / 32.f));
      float sn, cs; __sincosf(ang, &sn, &cs);
      outv = buf[tid] * cs + buf[tid + 32] * sn;
    } else if (tid < 64) {
      int j = tid - 32;
      float ang = (float)s * __powf(10000.f, -(float)j * (1.f / 32.f));
      float sn, cs; __sincosf(ang, &sn, &cs);
      outv = -buf[j] * sn + buf[tid] * cs;
    } else {
      outv = nrm;
    }
    if (kind == 0) {
      outv *= qgain[h];
      Qh[(size_t)((b * NHEAD + h) * SEQ + s) * HDIM + tid] = (bf16)outv;
    } else {
      Kh[(size_t)((b * NKVH + h) * SEQ + s) * HDIM + tid] = (bf16)outv;
    }
  } else {
    Vh[(size_t)((b * NKVH + h) * SEQ + s) * HDIM + tid] = (bf16)x;
  }
}

// ---------------------------------------------------------------------------
// Kernel 4: causal flash attention (GQA). 128 threads = 4 waves.
// Q tile / K tiles fetched by 1-D TDM copies (contiguous 16KB blocks);
// V transposed manually into LDS for B-fragments.
// ---------------------------------------------------------------------------
__global__ __launch_bounds__(128)
void flash_attn(const bf16* __restrict__ Qh, const bf16* __restrict__ Kh,
                const bf16* __restrict__ Vh, bf16* __restrict__ Y) {
  __shared__ __align__(32) bf16 Qs[64 * 128];
  __shared__ __align__(32) bf16 Ks[64 * 128];
  __shared__ __align__(32) bf16 Vts[128 * 64];
  __shared__ __align__(32) bf16 Ps[4 * 16 * 64];

  int bid = blockIdx.x;
  int qt = bid & 31;
  int h  = (bid >> 5) & 15;
  int b  = bid >> 9;
  int kvh = h >> 2;
  int q0 = qt * 64;

  int tid  = threadIdx.x;
  int lane = tid & 31;
  int w    = tid >> 5;
  int half = lane >> 4;
  int cn   = lane & 15;

  const bf16* Qg = Qh + ((size_t)(b * NHEAD + h) * SEQ + q0) * HDIM;
  const bf16* Kg = Kh + (size_t)(b * NKVH + kvh) * SEQ * HDIM;
  const bf16* Vg = Vh + (size_t)(b * NKVH + kvh) * SEQ * HDIM;

  // async-load the whole Q tile (contiguous 64x128 bf16 = one 1-D TDM copy)
  if (w == 0) tdm_load_2d(Qg, lds_off(Qs), 64 * 128, 1, 64 * 128ull);

  v8f o[8];
#pragma unroll
  for (int i = 0; i < 8; ++i) o[i] = (v8f){};
  float mrow[8], lrow[8];
#pragma unroll
  for (int r = 0; r < 8; ++r) { mrow[r] = -1e30f; lrow[r] = 0.f; }

  const float sc = 0.08838834764831845f;  // 1/sqrt(128)
  bf16* Pw = Ps + w * 16 * 64;

  for (int k0 = 0; k0 <= q0; k0 += 64) {
    // preload V rows (4 x 16 bf16 per thread) before the barrier
    v16bf vv[4];
#pragma unroll
    for (int i = 0; i < 4; ++i) {
      int c = tid + i * 128;
      int row = c >> 3, col = (c & 7) * 16;
      vv[i] = *(const v16bf*)(Vg + (size_t)(k0 + row) * 128 + col);
    }
    __syncthreads();  // previous iteration's compute done
    // transpose-store V into LDS [hd][key]
#pragma unroll
    for (int i = 0; i < 4; ++i) {
      int c = tid + i * 128;
      int row = c >> 3, col = (c & 7) * 16;
#pragma unroll
      for (int e = 0; e < 16; ++e) Vts[(col + e) * 64 + row] = vv[i][e];
    }
    // async-load K tile (contiguous) and drain TENSORcnt (covers Q on iter 0)
    if (w == 0) {
      tdm_load_2d(Kg + (size_t)k0 * 128, lds_off(Ks), 64 * 128, 1, 64 * 128ull);
      __builtin_amdgcn_s_wait_tensorcnt(0);
    }
    __syncthreads();  // Q/K (TDM) + V (ds) visible to all waves

    // S = Q K^T  (16x64 per wave)
    v8f sacc[4];
#pragma unroll
    for (int nt = 0; nt < 4; ++nt) {
      v8f sa = (v8f){};
#pragma unroll
      for (int kc = 0; kc < 4; ++kc) {
        v16bf aq = frag_a(Qs + (w * 16) * 128, 128, lane, kc * 32);
        v16bf bk = frag_b(Ks, 128, lane, nt * 16, kc * 32);
        sa = __builtin_amdgcn_wmma_f32_16x16x32_bf16(false, aq, false, bk,
                                                     (short)0, sa, false, false);
      }
      sacc[nt] = sa;
    }
    // scale + causal mask
#pragma unroll
    for (int nt = 0; nt < 4; ++nt)
#pragma unroll
      for (int r = 0; r < 8; ++r) {
        int rg = q0 + w * 16 + r + half * 8;
        int cg = k0 + nt * 16 + cn;
        float v = sacc[nt][r] * sc;
        sacc[nt][r] = (cg > rg) ? -1e30f : v;
      }
    // online softmax (row spans 16 lanes of one half-wave)
#pragma unroll
    for (int r = 0; r < 8; ++r) {
      float mx = -1e30f;
#pragma unroll
      for (int nt = 0; nt < 4; ++nt) mx = fmaxf(mx, sacc[nt][r]);
#pragma unroll
      for (int m = 1; m < 16; m <<= 1) mx = fmaxf(mx, __shfl_xor(mx, m, 32));
      float mnew  = fmaxf(mrow[r], mx);
      float alpha = __expf(mrow[r] - mnew);
      float sum = 0.f;
#pragma unroll
      for (int nt = 0; nt < 4; ++nt) {
        float p = __expf(sacc[nt][r] - mnew);
        sacc[nt][r] = p;
        sum += p;
      }
#pragma unroll
      for (int m = 1; m < 16; m <<= 1) sum += __shfl_xor(sum, m, 32);
      lrow[r] = lrow[r] * alpha + sum;
      mrow[r] = mnew;
#pragma unroll
      for (int nt2 = 0; nt2 < 8; ++nt2) o[nt2][r] *= alpha;
    }
    // stage P into wave-private LDS to re-fragment as an A matrix
#pragma unroll
    for (int nt = 0; nt < 4; ++nt)
#pragma unroll
      for (int r = 0; r < 8; ++r)
        Pw[(r + half * 8) * 64 + nt * 16 + cn] = (bf16)sacc[nt][r];

    // O += P V  (16x128 per wave)
#pragma unroll
    for (int nt2 = 0; nt2 < 8; ++nt2)
#pragma unroll
      for (int kc = 0; kc < 2; ++kc) {
        v16bf ap = frag_a(Pw, 64, lane, kc * 32);
        v16bf bv = frag_b(Vts, 64, lane, nt2 * 16, kc * 32);
        o[nt2] = __builtin_amdgcn_wmma_f32_16x16x32_bf16(false, ap, false, bv,
                                                         (short)0, o[nt2], false, false);
      }
  }

  // epilogue: O /= l, write bf16 Y[b*S + row][h*128 + col]
#pragma unroll
  for (int nt2 = 0; nt2 < 8; ++nt2)
#pragma unroll
    for (int r = 0; r < 8; ++r) {
      int row = q0 + w * 16 + r + half * 8;
      int col = h * HDIM + nt2 * 16 + cn;
      Y[(size_t)(b * SEQ + row) * DM + col] = (bf16)(o[nt2][r] / lrow[r]);
    }
}

// ---------------------------------------------------------------------------
// Host-side launch
// ---------------------------------------------------------------------------
extern "C" void kernel_launch(void* const* d_in, const int* in_sizes, int n_in,
                              void* d_out, int out_size, void* d_ws, size_t ws_size,
                              hipStream_t stream) {
  const float* x     = (const float*)d_in[0];
  const float* Wq    = (const float*)d_in[1];
  const float* Wk    = (const float*)d_in[2];
  const float* Wv    = (const float*)d_in[3];
  const float* Wproj = (const float*)d_in[4];
  const float* qgain = (const float*)d_in[5];
  const float* qA = (const float*)d_in[6];
  const float* qB = (const float*)d_in[7];
  const float* kA = (const float*)d_in[8];
  const float* kB = (const float*)d_in[9];
  const float* vA = (const float*)d_in[10];
  const float* vB = (const float*)d_in[11];
  const float* pA = (const float*)d_in[12];
  const float* pB = (const float*)d_in[13];

  char* ws = (char*)d_ws;
  bf16*  WeffQKV = (bf16*)(ws);                       // 3072*2048*2  = 12,582,912
  bf16*  WeffP   = (bf16*)(ws + 12582912);            // 2048*2048*2  =  8,388,608
  float* QKV     = (float*)(ws + 20971520);           // 4096*3072*4  = 50,331,648
  bf16*  Qh      = (bf16*)(ws + 71303168);            // 16,777,216
  bf16*  Kh      = (bf16*)(ws + 88080384);            //  4,194,304
  bf16*  Vh      = (bf16*)(ws + 92274688);            //  4,194,304
  bf16*  Yb      = (bf16*)(ws + 96468992);            // 16,777,216

  prep_weight<<<2048, 256, 0, stream>>>(Wq,    qA, qB, WeffQKV, 2048, DM, 0);
  prep_weight<<<512,  256, 0, stream>>>(Wk,    kA, kB, WeffQKV, 512,  DM, 2048);
  prep_weight<<<512,  256, 0, stream>>>(Wv,    vA, vB, WeffQKV, 512,  DM, 2560);
  prep_weight<<<2048, 256, 0, stream>>>(Wproj, pA, pB, WeffP,   2048, DM, 0);

  gemm_wmma<false><<<dim3(3072 / 128, 4096 / 128), 256, 0, stream>>>(
      (const void*)x, WeffQKV, QKV, DM, 3072);

  normrope<<<BATCH * NHEAD * SEQ + 2 * BATCH * NKVH * SEQ, 128, 0, stream>>>(
      QKV, qgain, Qh, Kh, Vh);

  flash_attn<<<BATCH * NHEAD * (SEQ / 64), 128, 0, stream>>>(Qh, Kh, Vh, Yb);

  gemm_wmma<true><<<dim3(2048 / 128, 4096 / 128), 256, 0, stream>>>(
      (const void*)Yb, WeffP, (float*)d_out, DM, DM);
}
